// GenerativeRNNmodel_82506321756339
// MI455X (gfx1250) — matrix operand order repaired and checked by reading, not compiled, not run on hardware
//
#include <hip/hip_runtime.h>

#define TLEN 2048
#define BATCH 64
#define HDIM 128
#define BT (BATCH * TLEN)

typedef _Float16 v16h __attribute__((ext_vector_type(16)));
typedef _Float16 h8   __attribute__((ext_vector_type(8)));
typedef float    v8f  __attribute__((ext_vector_type(8)));

union F16Frag { v16h v; h8 h[2]; };

// A fragment: 16x32 f16 tile, row-major source [m][k] with given stride.
// Layout (ISA 7.12.2): lane<16 -> elems0..7 = K kb..kb+7, elems8..15 = K kb+16..kb+23 (kb=k0)
//                      lane>=16 -> kb = k0+8.
__device__ __forceinline__ v16h load_a_frag(const _Float16* base, int stride, int m0, int k0, int lane) {
  const int m  = m0 + (lane & 15);
  const int kb = k0 + ((lane >> 4) << 3);
  const _Float16* p = base + m * stride + kb;
  F16Frag f;
  f.h[0] = *(const h8*)(p);
  f.h[1] = *(const h8*)(p + 16);
  return f.v;
}

// B fragment: 32x16 f16 tile where B[k][n] = src[n][k] (src row-major [n][k]).
// Lane owns column N=lane&15; lanes<16: elems0..7 = K k0..k0+7, elems8..15 = K k0+16..k0+23; lanes>=16: +8.
__device__ __forceinline__ v16h load_b_frag(const _Float16* base, int stride, int n0, int k0, int lane) {
  const int n  = n0 + (lane & 15);
  const int kb = k0 + ((lane >> 4) << 3);
  const _Float16* p = base + n * stride + kb;
  F16Frag f;
  f.h[0] = *(const h8*)(p);
  f.h[1] = *(const h8*)(p + 16);
  return f.v;
}

__device__ __forceinline__ v8f wmma_f16(v16h a, v16h b, v8f c) {
  return __builtin_amdgcn_wmma_f32_16x16x32_f16(false, a, false, b, (short)0, c, false, false);
}

__device__ __forceinline__ float lrelu(float v) { return v > 0.f ? v : 0.01f * v; }

// CDNA5 has a hardware TANH transcendental (v_tanh_f32). Use it directly, and
// build sigmoid from it: sigma(x) = 0.5*tanh(x/2) + 0.5 -- one TRANS op + FMA,
// no IEEE divide sequence on the recurrence critical path.
__device__ __forceinline__ float tanh_hw(float v) { return __builtin_amdgcn_tanhf(v); }
__device__ __forceinline__ float sigmoidf(float v) {
  return 0.5f + 0.5f * __builtin_amdgcn_tanhf(0.5f * v);
}

// CDNA5 async copy: 16 bytes global -> LDS, tracked with ASYNCcnt.
// ldsoff = LDS byte address (generic-address low 32 bits), gptr = global source.
__device__ __forceinline__ void async_copy_b128(unsigned ldsoff, const void* gptr) {
  asm volatile("global_load_async_to_lds_b128 %0, %1, off"
               :: "v"(ldsoff), "v"(gptr) : "memory");
}
__device__ __forceinline__ void wait_asynccnt0() {
  asm volatile("s_wait_asynccnt 0" ::: "memory");
}

// ---------------------------------------------------------------------------
// Kernel 0: convert fc weights to fp16 in workspace
// ---------------------------------------------------------------------------
__global__ void cvt_weights_kernel(const float* __restrict__ fc_w0,
                                   const float* __restrict__ fc_w1,
                                   _Float16* __restrict__ w0h,
                                   _Float16* __restrict__ w1h) {
  int i = blockIdx.x * blockDim.x + threadIdx.x;
  if (i < 256 * 256) {
    w0h[i] = (_Float16)fc_w0[i];
  } else {
    int j = i - 256 * 256;
    if (j < 64 * 256) w1h[j] = (_Float16)fc_w1[j];
  }
}

// ---------------------------------------------------------------------------
// Kernel 1: GRU scan. grid = 8 workgroups x 16 streams, block = 256 (8 waves).
// Streams 0..63 = forward time, 64..127 = reversed time.
// Loop-invariant A-fragments (w_hh, p_w0 tiles) are held in VGPRs across the
// whole 2048-step scan: at 8 waves/WGP there is ample VGPR headroom.
// ---------------------------------------------------------------------------
__global__ __launch_bounds__(256) void gru_scan_kernel(
    const float* __restrict__ x,
    const float* __restrict__ w_ih, const float* __restrict__ w_hh,
    const float* __restrict__ b_ih, const float* __restrict__ b_hh,
    const float* __restrict__ p_w0, const float* __restrict__ p_b0,
    const float* __restrict__ p_w1, const float* __restrict__ p_b1,
    const float* __restrict__ p_w2, const float* __restrict__ p_b2,
    _Float16* __restrict__ feats, float* __restrict__ preds_out)
{
  extern __shared__ char smem[];
  _Float16* whh  = (_Float16*)smem;          // 384*128 fp16 (96 KB)
  _Float16* hbuf = whh + 384 * 128;          // 2 x 16*128 fp16 ping-pong
  _Float16* pw0  = hbuf + 2 * 16 * 128;      // 64*128 fp16
  float* wih      = (float*)(pw0 + 64 * 128);// 384
  float* bihs     = wih + 384;               // 384
  float* bhhs     = bihs + 384;              // 384
  float* pb0      = bhhs + 384;              // 64
  float* pw1      = pb0 + 64;                // 64
  float* xin      = pw1 + 64;                // 16
  float* predprev = xin + 16;                // 16
  float* a2       = predprev + 16;           // 16
  float* sc       = a2 + 16;                 // 3 scalars: p_b1, p_w2, p_b2

  const int tid  = threadIdx.x;
  const int lane = tid & 31;
  const int wave = tid >> 5;
  const int s0   = blockIdx.x * 16;          // first global stream of this group

  // ---- init: stage weights into LDS (fp16 where they feed WMMA) ----
  for (int i = tid; i < 384 * 128; i += 256) whh[i] = (_Float16)w_hh[i];
  for (int i = tid; i < 64 * 128;  i += 256) pw0[i] = (_Float16)p_w0[i];
  for (int i = tid; i < 384; i += 256) { wih[i] = w_ih[i]; bihs[i] = b_ih[i]; bhhs[i] = b_hh[i]; }
  if (tid < 64) { pb0[tid] = p_b0[tid]; pw1[tid] = p_w1[tid]; }
  if (tid == 0) { sc[0] = p_b1[0]; sc[1] = p_w2[0]; sc[2] = p_b2[0]; }
  for (int i = tid; i < 16 * 128; i += 256) hbuf[i] = (_Float16)0.f;  // h0 = 0 (buffer 0)
  __syncthreads();

  // ---- hoist loop-invariant A-fragments into registers ----
  const int kt = wave;                       // each wave owns 16 hidden dims
  v16h aR[4], aZ[4], aN[4];
  #pragma unroll
  for (int ks = 0; ks < 4; ++ks) {
    aR[ks] = load_a_frag(whh, 128, kt * 16,       ks * 32, lane);
    aZ[ks] = load_a_frag(whh, 128, 128 + kt * 16, ks * 32, lane);
    aN[ks] = load_a_frag(whh, 128, 256 + kt * 16, ks * 32, lane);
  }
  v16h aP[4];
  if (wave < 4) {
    #pragma unroll
    for (int ks = 0; ks < 4; ++ks) aP[ks] = load_a_frag(pw0, 128, wave * 16, ks * 32, lane);
  }

  for (int t = 0; t < TLEN; ++t) {
    const int cur = t & 1;
    const _Float16* hc = hbuf + cur * (16 * 128);
    _Float16*       hn = hbuf + (cur ^ 1) * (16 * 128);

    // ---- P1: gather x (with NaN imputation), record preds ----
    if (tid < 16) {
      const int s   = s0 + tid;
      const int rev = (s >= BATCH);
      const int b   = rev ? (s - BATCH) : s;
      const int tt  = rev ? (TLEN - 1 - t) : t;
      float xv = x[b * TLEN + tt];
      float pv = predprev[tid];
      float used = (t == 0) ? xv : pv;              // preds[0]=raw first input, else pred_prev
      float xi = xv;
      if (t > 0 && (xv != xv)) xi = pv;             // NaN -> previous prediction
      xin[tid] = xi;
      preds_out[s * TLEN + (rev ? (TLEN - 1 - t) : t)] = used;
    } else if (tid < 32) {
      a2[tid - 16] = 0.f;
    }
    __syncthreads();

    // ---- P2: gh = h @ w_hh^T (WMMA), fused GRU gates ----
    {
      v8f accR = {}; v8f accZ = {}; v8f accN = {};
      #pragma unroll
      for (int ks = 0; ks < 4; ++ks) {
        v16h bfr = load_b_frag(hc, 128, 0, ks * 32, lane);   // shared across r/z/n
        accR = wmma_f16(aR[ks], bfr, accR);
        accZ = wmma_f16(aZ[ks], bfr, accZ);
        accN = wmma_f16(aN[ks], bfr, accN);
      }
      const int nloc = lane & 15;                   // stream within group (D-tile N)
      const int kb   = kt * 16 + ((lane >> 4) << 3);// 8 consecutive hidden dims (D-tile M)
      const float xi = xin[nloc];
      const h8 hold8 = *(const h8*)(hc + nloc * 128 + kb);
      h8 hnew8;
      #pragma unroll
      for (int r = 0; r < 8; ++r) {
        const int k = kb + r;
        float gr = accR[r] + xi * wih[k]       + bihs[k]       + bhhs[k];
        float gz = accZ[r] + xi * wih[128 + k] + bihs[128 + k] + bhhs[128 + k];
        float rg = sigmoidf(gr);
        float zg = sigmoidf(gz);
        float nh = accN[r] + bhhs[256 + k];         // gh_n (+b_hh)
        float ni = xi * wih[256 + k] + bihs[256 + k];
        float ng = tanh_hw(ni + rg * nh);
        float ho = (float)hold8[r];
        hnew8[r] = (_Float16)((1.f - zg) * ng + zg * ho);
      }
      *(h8*)(hn + nloc * 128 + kb) = hnew8;
      // bidirectional feature recombination: feats[b][t] = [h_fwd(t) | h_bwd(T-1-t)]
      const int s = s0 + nloc;
      long long foff;
      if (s < BATCH) foff = ((long long)s * TLEN + t) * 256 + kb;
      else           foff = ((long long)(s - BATCH) * TLEN + (TLEN - 1 - t)) * 256 + 128 + kb;
      *(h8*)(feats + foff) = hnew8;
    }
    __syncthreads();

    // ---- P3: pred MLP layer0 (WMMA, M=64) + fused layer1 dot-reduce ----
    if (wave < 4) {
      v8f acc = {};
      #pragma unroll
      for (int ks = 0; ks < 4; ++ks) {
        v16h bf = load_b_frag(hn, 128, 0, ks * 32, lane);
        acc = wmma_f16(aP[ks], bf, acc);
      }
      const int nloc = lane & 15;
      const int mb   = wave * 16 + ((lane >> 4) << 3);
      float part = 0.f;
      #pragma unroll
      for (int r = 0; r < 8; ++r) {
        const int m = mb + r;
        part += lrelu(acc[r] + pb0[m]) * pw1[m];
      }
      atomicAdd(&a2[nloc], part);                   // ds_add_f32
    }
    __syncthreads();

    // ---- P4: pred head -> pred_prev for next step ----
    if (tid < 16) {
      predprev[tid] = lrelu(a2[tid] + sc[0]) * sc[1] + sc[2];
    }
    __syncthreads();
  }
}

// ---------------------------------------------------------------------------
// Kernel 2: fc MLP 256->256->64->1 + sigmoid over B*T rows.
// grid = BT/128 workgroups, block = 512 (16 waves), 128 rows per workgroup.
// The 128x256 fp16 feature tile is staged into LDS once via the CDNA5
// async-copy path (global_load_async_to_lds_b128 / ASYNCcnt), so each row is
// fetched from memory once instead of once per m-tile.
// ---------------------------------------------------------------------------
__global__ __launch_bounds__(512) void fc_mlp_kernel(
    const _Float16* __restrict__ feats,
    const _Float16* __restrict__ w0h, const _Float16* __restrict__ w1h,
    const float* __restrict__ fc_b0, const float* __restrict__ fc_b1,
    const float* __restrict__ fc_w2, const float* __restrict__ fc_b2,
    float* __restrict__ out)
{
  extern __shared__ char smem[];
  _Float16* ftile = (_Float16*)smem;    // 128 x 256 fp16 (64 KB)
  _Float16* act1  = ftile + 128 * 256;  // 128 x 256 fp16 (64 KB)
  _Float16* act2  = act1 + 128 * 256;   // 128 x 64 fp16 (16 KB)

  const int tid  = threadIdx.x;
  const int lane = tid & 31;
  const int wave = tid >> 5;
  const long long r0 = (long long)blockIdx.x * 128;
  const _Float16* fblk = feats + r0 * 256;

  // ---- async-stage the contiguous 64 KB feature tile into LDS ----
  {
    const unsigned ftile_base = (unsigned)(uintptr_t)(void*)ftile;  // LDS byte offset
    #pragma unroll
    for (int j = 0; j < 8; ++j) {
      const int c = j * 512 + tid;                  // 4096 x 16B chunks
      async_copy_b128(ftile_base + (unsigned)c * 16u, (const void*)(fblk + c * 8));
    }
    wait_asynccnt0();
  }

  // hoist per-wave layer-0 A-fragments (mt = wave is fixed for this wave)
  v16h a0[8];
  #pragma unroll
  for (int ks = 0; ks < 8; ++ks) a0[ks] = load_a_frag(w0h, 256, wave * 16, ks * 32, lane);
  __syncthreads();                                  // ftile visible to all waves

  // layer 0: out 256 <- in 256, leaky relu. wave = m-tile, nt = 0..7.
  #pragma unroll
  for (int nt = 0; nt < 8; ++nt) {
    v8f acc = {};
    #pragma unroll
    for (int ks = 0; ks < 8; ++ks) {
      acc = wmma_f16(a0[ks], load_b_frag(ftile, 256, nt * 16, ks * 32, lane), acc);
    }
    const int nloc = nt * 16 + (lane & 15);
    const int mb   = wave * 16 + ((lane >> 4) << 3);
    h8 st;
    #pragma unroll
    for (int r = 0; r < 8; ++r) st[r] = (_Float16)lrelu(acc[r] + fc_b0[mb + r]);
    *(h8*)(act1 + nloc * 256 + mb) = st;
  }
  __syncthreads();

  // layer 1: out 64 <- in 256, leaky relu. mt = wave&3; nt = wave>>2 and +4.
  {
    const int mt = wave & 3;
    v16h a1f[8];
    #pragma unroll
    for (int ks = 0; ks < 8; ++ks) a1f[ks] = load_a_frag(w1h, 256, mt * 16, ks * 32, lane);
    #pragma unroll
    for (int j = 0; j < 2; ++j) {
      const int nt = (wave >> 2) + 4 * j;
      v8f acc = {};
      #pragma unroll
      for (int ks = 0; ks < 8; ++ks) {
        acc = wmma_f16(a1f[ks], load_b_frag(act1, 256, nt * 16, ks * 32, lane), acc);
      }
      const int nloc = nt * 16 + (lane & 15);
      const int mb   = mt * 16 + ((lane >> 4) << 3);
      h8 st;
      #pragma unroll
      for (int r = 0; r < 8; ++r) st[r] = (_Float16)lrelu(acc[r] + fc_b1[mb + r]);
      *(h8*)(act2 + nloc * 64 + mb) = st;
    }
  }
  __syncthreads();

  // layer 2: 64 -> 1, then sigmoid. One thread per row.
  if (tid < 128) {
    float acc = fc_b2[0];
    #pragma unroll
    for (int k = 0; k < 64; ++k) acc += (float)act2[tid * 64 + k] * fc_w2[k];
    const long long row = r0 + tid;
    out[row] = sigmoidf(acc);      // probs (0.5*tanh(x/2)+0.5)
    out[BT + row] = acc;           // logits
  }
}

// ---------------------------------------------------------------------------
extern "C" void kernel_launch(void* const* d_in, const int* in_sizes, int n_in,
                              void* d_out, int out_size, void* d_ws, size_t ws_size,
                              hipStream_t stream) {
  (void)in_sizes; (void)n_in; (void)out_size; (void)ws_size;
  const float* x     = (const float*)d_in[0];
  const float* w_ih  = (const float*)d_in[1];
  const float* w_hh  = (const float*)d_in[2];
  const float* b_ih  = (const float*)d_in[3];
  const float* b_hh  = (const float*)d_in[4];
  const float* fc_w0 = (const float*)d_in[5];
  const float* fc_b0 = (const float*)d_in[6];
  const float* fc_w1 = (const float*)d_in[7];
  const float* fc_b1 = (const float*)d_in[8];
  const float* fc_w2 = (const float*)d_in[9];
  const float* fc_b2 = (const float*)d_in[10];
  const float* p_w0  = (const float*)d_in[11];
  const float* p_b0  = (const float*)d_in[12];
  const float* p_w1  = (const float*)d_in[13];
  const float* p_b1  = (const float*)d_in[14];
  const float* p_w2  = (const float*)d_in[15];
  const float* p_b2  = (const float*)d_in[16];
  float* out = (float*)d_out;

  char* ws = (char*)d_ws;
  _Float16* feats = (_Float16*)ws;                                   // BT x 256 fp16 (64 MB)
  _Float16* w0h   = (_Float16*)(ws + (size_t)BT * 256 * sizeof(_Float16));
  _Float16* w1h   = w0h + 256 * 256;

  cvt_weights_kernel<<<320, 256, 0, stream>>>(fc_w0, fc_w1, w0h, w1h);

  const size_t gruLds = (size_t)(384 * 128 + 2 * 16 * 128 + 64 * 128) * sizeof(_Float16)
                      + (size_t)(384 * 3 + 64 * 2 + 16 * 3 + 4) * sizeof(float);
  gru_scan_kernel<<<8, 256, gruLds, stream>>>(x, w_ih, w_hh, b_ih, b_hh,
      p_w0, p_b0, p_w1, p_b1, p_w2, p_b2, feats, out + 2 * BT);

  const size_t fcLds = (size_t)(128 * 256 + 128 * 256 + 128 * 64) * sizeof(_Float16);
  fc_mlp_kernel<<<BT / 128, 512, fcLds, stream>>>(feats, w0h, w1h,
      fc_b0, fc_b1, fc_w2, fc_b2, out);
}
